// EdgeConv_39084202394050
// MI455X (gfx1250) — matrix operand order repaired and checked by reading, not compiled, not run on hardware
//
#include <hip/hip_runtime.h>

#define BB 64
#define NN 4096
#define DD 128
#define HH 256

typedef __attribute__((ext_vector_type(16))) __bf16 v16bf;
typedef __attribute__((ext_vector_type(8)))  float  v8f;

static __device__ __forceinline__ __bf16 f2bf(float f) {
  unsigned u = __builtin_bit_cast(unsigned, f);
  unsigned r = (u + 0x7FFFu + ((u >> 16) & 1u)) >> 16;   // RNE fp32->bf16
  unsigned short s = (unsigned short)r;
  return __builtin_bit_cast(__bf16, s);
}
static __device__ __forceinline__ float bf2f(__bf16 b) {
  unsigned u = ((unsigned)__builtin_bit_cast(unsigned short, b)) << 16;
  return __builtin_bit_cast(float, u);
}

// ---------------------------------------------------------------------------
// K0: split W1 (fp32 [D,H]) into bf16 hi/lo, pre-swizzled into this code's
// contiguous-K B-operand layout: blob[plane][ht][k][lane][j], where lane = column
// (h = ht*16 + lane%16) and halfs j=0..15 are K = k*32 + (lane>=16?16:0) + j.
// ---------------------------------------------------------------------------
__global__ void prep_kernel(const float* __restrict__ W1,
                            unsigned short* __restrict__ blob) {
  int e = blockIdx.x * 256 + threadIdx.x;   // 0 .. 32767
  int j    = e & 15;
  int lane = (e >> 4) & 31;
  int k    = (e >> 9) & 3;
  int ht   = e >> 11;
  int d = k * 32 + (lane >> 4) * 16 + j;
  int h = ht * 16 + (lane & 15);
  float v = W1[d * HH + h];
  __bf16 hi = f2bf(v);
  __bf16 lo = f2bf(v - bf2f(hi));
  blob[e]         = __builtin_bit_cast(unsigned short, hi);
  blob[32768 + e] = __builtin_bit_cast(unsigned short, lo);
}

// ---------------------------------------------------------------------------
// K1: fused scores = relu(feats@W1 + b1)@W2 + b2 using bf16x3 WMMA.
// 512 WGs x 256 thr (8 waves). Each WG handles 512 nodes (4 tiles/wave).
// W1 blob (128 KB) is async-copied global->LDS once per WG.
// ---------------------------------------------------------------------------
__global__ void __launch_bounds__(256)
score_kernel(const float* __restrict__ feats,
             const unsigned short* __restrict__ blob,
             const float* __restrict__ b1,
             const float* __restrict__ W2,
             const float* __restrict__ b2,
             float* __restrict__ scores) {
  extern __shared__ unsigned short sblob[];   // 131072 bytes: [2][16][4][32][16]
  const int t = threadIdx.x;

  // Async copy 128 KB: 8192 x b128 chunks, 32 per thread (ASYNCcnt-tracked).
  for (int i = 0; i < 32; ++i) {
    unsigned off = (unsigned)((i * 256 + t) * 16);
    unsigned lds = (unsigned)(size_t)((char*)sblob + off);          // low 32 = LDS addr
    unsigned long long g = (unsigned long long)(const char*)blob + off;
    asm volatile("global_load_async_to_lds_b128 %0, %1, off"
                 :: "v"(lds), "v"(g) : "memory");
  }
  asm volatile("s_wait_asynccnt 0x0" ::: "memory");
  __syncthreads();

  const int wave = t >> 5, lane = t & 31, row = lane & 15, hs = lane >> 4;
  const int b = blockIdx.x >> 3;              // 8 node-blocks of 512 per batch
  const int nodeBlk = (blockIdx.x & 7) * 512;
  const float b2v = b2[0];
  const unsigned short* shi = sblob;
  const unsigned short* slo = sblob + 32768;

  for (int tile = 0; tile < 4; ++tile) {
    const int tbase = nodeBlk + (wave * 4 + tile) * 16;
    const int node  = tbase + row;
    const float* fr = feats + ((size_t)b * NN + node) * DD;

    // A-matrix (16x32 bf16 per K-step): lane holds row M=lane%16;
    // lanes 0-15: K 0-15, lanes 16-31: K 16-31 (per K-block).
    v16bf ah[4], al[4];
#pragma unroll
    for (int k = 0; k < 4; ++k) {
      const float* p = fr + k * 32 + hs * 16;
      float4 x0 = *(const float4*)(p);
      float4 x1 = *(const float4*)(p + 4);
      float4 y0 = *(const float4*)(p + 8);
      float4 y1 = *(const float4*)(p + 12);
      float f[16] = {x0.x, x0.y, x0.z, x0.w, x1.x, x1.y, x1.z, x1.w,
                     y0.x, y0.y, y0.z, y0.w, y1.x, y1.y, y1.z, y1.w};
#pragma unroll
      for (int j = 0; j < 16; ++j) {
        __bf16 h = f2bf(f[j]);
        ah[k][j] = h;
        al[k][j] = f2bf(f[j] - bf2f(h));
      }
    }

    float sacc[8] = {0.f, 0.f, 0.f, 0.f, 0.f, 0.f, 0.f, 0.f};
    for (int ht = 0; ht < 16; ++ht) {
      v8f c = {0.f, 0.f, 0.f, 0.f, 0.f, 0.f, 0.f, 0.f};
#pragma unroll
      for (int k = 0; k < 4; ++k) {
        unsigned idx = ((unsigned)(ht * 4 + k) * 32 + (unsigned)lane) * 16u;
        v16bf bh = *(const v16bf*)(shi + idx);
        v16bf bl = *(const v16bf*)(slo + idx);
        // fp32-faithful: hi*hi + lo*hi + hi*lo (drop lo*lo, ~2^-18 relative)
        c = __builtin_amdgcn_wmma_f32_16x16x32_bf16(false, ah[k], false, bh,
                                                    (short)0, c, false, false);
        c = __builtin_amdgcn_wmma_f32_16x16x32_bf16(false, al[k], false, bh,
                                                    (short)0, c, false, false);
        c = __builtin_amdgcn_wmma_f32_16x16x32_bf16(false, ah[k], false, bl,
                                                    (short)0, c, false, false);
      }
      // Epilogue: D tile lane holds column h = ht*16 + lane%16, rows r (+8 if hs).
      const int h = ht * 16 + row;
      const float w2v = W2[h], b1v = b1[h];
#pragma unroll
      for (int r = 0; r < 8; ++r) {
        float x = c[r] + b1v;
        x = x > 0.f ? x : 0.f;
        sacc[r] += x * w2v;
      }
    }
    // Sum over the 16 columns: reduce within each 16-lane half (wave32).
#pragma unroll
    for (int ms = 1; ms <= 8; ms <<= 1)
#pragma unroll
      for (int r = 0; r < 8; ++r)
        sacc[r] += __shfl_xor(sacc[r], ms, 32);

    if (lane == 0) {
#pragma unroll
      for (int r = 0; r < 8; ++r)
        scores[(size_t)b * NN + tbase + r] = sacc[r] + b2v;
    } else if (lane == 16) {
#pragma unroll
      for (int r = 0; r < 8; ++r)
        scores[(size_t)b * NN + tbase + 8 + r] = sacc[r] + b2v;
    }
  }
}

// ---------------------------------------------------------------------------
// K2: per-batch softmax stats (max, sum-of-exp). 64 WGs x 256 thr.
// ---------------------------------------------------------------------------
__global__ void stats_kernel(const float* __restrict__ scores,
                             float* __restrict__ stats) {
  const int b = blockIdx.x, t = threadIdx.x;
  __shared__ float red[8];
  const float* s = scores + (size_t)b * NN;

  float m = -3.4e38f;
  for (int i = t; i < NN; i += 256) m = fmaxf(m, s[i]);
#pragma unroll
  for (int k = 1; k < 32; k <<= 1) m = fmaxf(m, __shfl_xor(m, k, 32));
  if ((t & 31) == 0) red[t >> 5] = m;
  __syncthreads();
  float mg = red[0];
#pragma unroll
  for (int w = 1; w < 8; ++w) mg = fmaxf(mg, red[w]);
  __syncthreads();

  float l = 0.f;
  for (int i = t; i < NN; i += 256) l += expf(s[i] - mg);
#pragma unroll
  for (int k = 1; k < 32; k <<= 1) l += __shfl_xor(l, k, 32);
  if ((t & 31) == 0) red[t >> 5] = l;
  __syncthreads();
  if (t == 0) {
    float lg = 0.f;
#pragma unroll
    for (int w = 0; w < 8; ++w) lg += red[w];
    stats[2 * b]     = mg;
    stats[2 * b + 1] = lg;
  }
}

// ---------------------------------------------------------------------------
// K3: partial weighted sums. 1024 WGs (64 batches x 16 node-chunks of 256).
// Deterministic (no float atomics): partials[b][chunk][d].
// ---------------------------------------------------------------------------
__global__ void wsum_kernel(const float* __restrict__ feats,
                            const float* __restrict__ scores,
                            const float* __restrict__ stats,
                            float* __restrict__ partials) {
  const int blk = blockIdx.x, b = blk >> 4, c = blk & 15, t = threadIdx.x;
  __shared__ float wbuf[256];
  __shared__ float acc[256];
  const float m = stats[2 * b], l = stats[2 * b + 1];
  const int nbase = c * 256;
  wbuf[t] = expf(scores[(size_t)b * NN + nbase + t] - m) / l;
  __syncthreads();

  const int d = t & 127, g = t >> 7;          // two groups of 128 nodes
  const float* fp = feats + ((size_t)b * NN + nbase + g * 128) * DD + d;
  float a = 0.f;
  for (int i = 0; i < 128; ++i) a += wbuf[g * 128 + i] * fp[(size_t)i * DD];
  acc[t] = a;
  __syncthreads();
  if (t < 128)
    partials[((size_t)b * 16 + c) * DD + t] = acc[t] + acc[t + 128];
}

// K4: fold 16 chunk-partials into d_out. 64 WGs x 128 thr.
__global__ void reduce_kernel(const float* __restrict__ partials,
                              float* __restrict__ out) {
  const int b = blockIdx.x, t = threadIdx.x;
  float a = 0.f;
#pragma unroll
  for (int c = 0; c < 16; ++c) a += partials[((size_t)b * 16 + c) * DD + t];
  out[b * DD + t] = a;
}

extern "C" void kernel_launch(void* const* d_in, const int* in_sizes, int n_in,
                              void* d_out, int out_size, void* d_ws, size_t ws_size,
                              hipStream_t stream) {
  const float* feats = (const float*)d_in[0];
  const float* W1    = (const float*)d_in[1];
  const float* b1    = (const float*)d_in[2];
  const float* W2    = (const float*)d_in[3];
  const float* b2    = (const float*)d_in[4];
  float* out = (float*)d_out;

  // d_ws layout (floats): scores[B*N] | stats[256] | partials[B*16*128] | W1 blob
  float* ws       = (float*)d_ws;
  float* scores   = ws;                               // 262144 floats
  float* stats    = ws + (size_t)BB * NN;             // 256 floats (128 used)
  float* partials = stats + 256;                      // 131072 floats
  unsigned short* w1blob = (unsigned short*)(partials + (size_t)BB * 16 * DD);

  prep_kernel<<<128, 256, 0, stream>>>(W1, w1blob);
  score_kernel<<<512, 256, 131072, stream>>>(feats, w1blob, b1, W2, b2, scores);
  stats_kernel<<<BB, 256, 0, stream>>>(scores, stats);
  wsum_kernel<<<BB * 16, 256, 0, stream>>>(feats, scores, stats, partials);
  reduce_kernel<<<BB, 128, 0, stream>>>(partials, out);
}
